// LoRALinear_27504970564261
// MI455X (gfx1250) — matrix operand order repaired and checked by reading, not compiled, not run on hardware
//
#include <hip/hip_runtime.h>
#include <hip/hip_bf16.h>
#include <stdint.h>

typedef __attribute__((ext_vector_type(16))) __bf16 v16bf;
typedef __attribute__((ext_vector_type(8)))  __bf16 bf16x8;
typedef __attribute__((ext_vector_type(4)))  __bf16 bf16x4;
typedef __attribute__((ext_vector_type(8)))  float  v8f;
typedef int v4i_g __attribute__((vector_size(16)));   // GCC-vector int4, per builtin signature

#define BQ 8
#define SQ 2048
#define INQ 4096
#define OUTQ 4096
#define RANK 64
#define MTOT (BQ * SQ)   // 16384
#define STAGES 4
#define TM 256            // block tile M
#define TN 128            // block tile N

#if __has_builtin(__builtin_amdgcn_global_load_async_to_lds_b128)
#define HAVE_ASYNC_LDS 1
#endif

union FragU { v16bf v; bf16x8 h[2]; };

// A-matrix fragment (16x32 bf16), rows m = l16, per ISA layout:
// lanes 0-15: K 0-7 in VGPR0-3, K 16-23 in VGPR4-7; lanes 16-31: K 8-15, 24-31.
__device__ __forceinline__ v16bf load_fragA(const __bf16* base, int stride,
                                            int l16, int half) {
    const __bf16* p = base + (size_t)l16 * stride + half * 8;
    FragU u;
    u.h[0] = *(const bf16x8*)(p);
    u.h[1] = *(const bf16x8*)(p + 16);
    return u.v;
}

// B-matrix fragment (32x16 bf16), stored [n][k] k-contiguous; column n = l16,
// lanes 0-15: K 0-7, 16-23; lanes 16-31: K 8-15, 24-31 (same as A).
__device__ __forceinline__ v16bf load_fragB(const __bf16* base, int stride,
                                            int l16, int half) {
    const __bf16* p = base + (size_t)l16 * stride + half * 8;
    FragU u;
    u.h[0] = *(const bf16x8*)(p);
    u.h[1] = *(const bf16x8*)(p + 16);
    return u.v;
}

__device__ __forceinline__ v8f wmma_bf16(v16bf a, v16bf b, v8f c) {
    return __builtin_amdgcn_wmma_f32_16x16x32_bf16(
        false, a, false, b, (short)0, c, false, false);
}

#ifdef HAVE_ASYNC_LDS
// Copy 32B per lane: two b128 async copies; the IOFFSET applies to both the
// global and LDS addresses per ISA GLOBAL_LOAD_ASYNC_TO_LDS semantics.
__device__ __forceinline__ void async_cp32(const __bf16* g, __bf16* l) {
    auto gp = (__attribute__((address_space(1))) v4i_g*)(uintptr_t)g;
    auto lp = (__attribute__((address_space(3))) v4i_g*)(uintptr_t)l;
    __builtin_amdgcn_global_load_async_to_lds_b128(gp, lp, 0, 0);
    __builtin_amdgcn_global_load_async_to_lds_b128(gp, lp, 16, 0);
}
template <int N>
__device__ __forceinline__ void wait_async() {
#if __has_builtin(__builtin_amdgcn_s_wait_asynccnt)
    __builtin_amdgcn_s_wait_asynccnt(N);
#else
    asm volatile("s_wait_asynccnt %0" :: "i"(N) : "memory");
#endif
}
#endif

// ---------------------------------------------------------------- converts
__global__ __launch_bounds__(256) void convert_f32_bf16(
    const float* __restrict__ in, __bf16* __restrict__ out) {
    size_t i = ((size_t)blockIdx.x * 256 + threadIdx.x) * 4;
    float4 f = *(const float4*)(in + i);
    bf16x4 b;
    b.x = (__bf16)f.x; b.y = (__bf16)f.y; b.z = (__bf16)f.z; b.w = (__bf16)f.w;
    *(bf16x4*)(out + i) = b;
}

// lora_a: [8][IN][RANK] -> abT: [8][RANK][IN]  (k-contiguous rows)
__global__ __launch_bounds__(256) void transpose_a(
    const float* __restrict__ a, __bf16* __restrict__ abT) {
    size_t idx = (size_t)blockIdx.x * 256 + threadIdx.x;
    size_t ad  = idx / ((size_t)RANK * INQ);
    size_t rem = idx % ((size_t)RANK * INQ);
    size_t r = rem / INQ, k = rem % INQ;
    abT[idx] = (__bf16)a[ad * ((size_t)INQ * RANK) + k * RANK + r];
}

// lora_b: [8][RANK][OUT] -> bbT: [8][OUT][RANK]  (r-contiguous rows)
__global__ __launch_bounds__(256) void transpose_b(
    const float* __restrict__ bw, __bf16* __restrict__ bbT) {
    size_t idx = (size_t)blockIdx.x * 256 + threadIdx.x;
    size_t ad  = idx / ((size_t)OUTQ * RANK);
    size_t rem = idx % ((size_t)OUTQ * RANK);
    size_t o = rem / RANK, r = rem % RANK;
    bbT[idx] = (__bf16)bw[ad * ((size_t)RANK * OUTQ) + r * OUTQ + o];
}

// ---------------------------------------------------------------- shrink
__global__ __launch_bounds__(256) void lora_shrink(
    const __bf16* __restrict__ xb, const __bf16* __restrict__ abT,
    const int* __restrict__ lora_id, __bf16* __restrict__ temp) {
    int tid = threadIdx.x;
    int wave = tid >> 5, lane = tid & 31;
    int l16 = lane & 15, half = lane >> 4;
    size_t M0 = (size_t)blockIdx.x * 128;
    int aid = lora_id[M0 / SQ];

    const __bf16* Ap = xb + (M0 + (size_t)wave * 16) * INQ;
    const __bf16* Bp = abT + (size_t)aid * RANK * INQ;

    v8f acc[4];
#pragma unroll
    for (int j = 0; j < 4; ++j) acc[j] = (v8f)(0.0f);

    for (int kt = 0; kt < INQ / 32; ++kt) {
        v16bf af = load_fragA(Ap + kt * 32, INQ, l16, half);
#pragma unroll
        for (int j = 0; j < 4; ++j) {
            v16bf bf_ = load_fragB(Bp + (size_t)(j * 16) * INQ + kt * 32,
                                   INQ, l16, half);
            acc[j] = wmma_bf16(af, bf_, acc[j]);
        }
    }
#pragma unroll
    for (int j = 0; j < 4; ++j) {
        size_t m = M0 + wave * 16 + half * 8;
#pragma unroll
        for (int v = 0; v < 8; ++v)
            temp[(m + v) * RANK + j * 16 + l16] = (__bf16)acc[j][v];
    }
}

// ---------------------------------------------------------------- main GEMM
// out[m][n] = bias[n] + xb[m]·wb[n] + temp[m]·bbT[id][n]
// 256x128 block tile, 8 waves of 64x64 (16 WMMA / K-step / wave), K-step 32,
// STAGES-deep async-LDS pipeline (6 async instrs / stage / wave).
__global__ __launch_bounds__(256) void lora_main(
    const __bf16* __restrict__ xb, const __bf16* __restrict__ wb,
    const float* __restrict__ bias, const __bf16* __restrict__ temp,
    const __bf16* __restrict__ bbT, const int* __restrict__ lora_id,
    float* __restrict__ out) {
    __shared__ __bf16 lA[STAGES][TM * 32];   // 4 x 16 KB
    __shared__ __bf16 lB[STAGES][TN * 32];   // 4 x  8 KB

    int tid = threadIdx.x;
    int wave = tid >> 5, lane = tid & 31;
    int l16 = lane & 15, half = lane >> 4;
    int waveM = (wave & 3) * 64;   // 4 waves over M
    int waveN = (wave >> 2) * 64;  // 2 waves over N

    size_t M0 = (size_t)blockIdx.x * TM;
    size_t N0 = (size_t)blockIdx.y * TN;

    const __bf16* xrow = xb + M0 * INQ;
    const __bf16* wrow = wb + N0 * INQ;

    // staging assignment: A: thread t owns row t (64B); B: thread t owns half
    // of row t>>1 (32B). 6 async instructions per wave per stage.
    const __bf16* gA0 = xrow + (size_t)tid * INQ;
    int bRow = tid >> 1, bSeg = (tid & 1) * 16;
    const __bf16* gB0 = wrow + (size_t)bRow * INQ + bSeg;

    v8f acc[4][4];
#pragma unroll
    for (int i = 0; i < 4; ++i)
#pragma unroll
        for (int j = 0; j < 4; ++j) acc[i][j] = (v8f)(0.0f);

    const int KT = INQ / 32;  // 128

#ifdef HAVE_ASYNC_LDS
    auto issue = [&](int kt, int s) {
        const __bf16* ga = gA0 + (size_t)kt * 32;
        async_cp32(ga,      &lA[s][tid * 32]);
        async_cp32(ga + 16, &lA[s][tid * 32 + 16]);
        async_cp32(gB0 + (size_t)kt * 32, &lB[s][bRow * 32 + bSeg]);
    };
#pragma unroll
    for (int s = 0; s < STAGES - 1; ++s) issue(s, s);

    for (int kt = 0; kt < KT; ++kt) {
        // wait until own stage-kt copies land; threshold = 6 * stages-ahead
        int rem = KT - 1 - kt;
        if (rem >= 2)      wait_async<12>();
        else if (rem == 1) wait_async<6>();
        else               wait_async<0>();
        __syncthreads();   // all waves' stage kt in LDS; prior reads of the
                           // reuse buffer retired
        if (kt + STAGES - 1 < KT) issue(kt + STAGES - 1, (kt + STAGES - 1) % STAGES);

        int cur = kt % STAGES;
        v16bf afrag[4], bfrag[4];
#pragma unroll
        for (int i = 0; i < 4; ++i)
            afrag[i] = load_fragA(&lA[cur][(waveM + i * 16) * 32], 32, l16, half);
#pragma unroll
        for (int j = 0; j < 4; ++j)
            bfrag[j] = load_fragB(&lB[cur][(waveN + j * 16) * 32], 32, l16, half);
#pragma unroll
        for (int i = 0; i < 4; ++i)
#pragma unroll
            for (int j = 0; j < 4; ++j)
                acc[i][j] = wmma_bf16(afrag[i], bfrag[j], acc[i][j]);
    }
#else
    // ---- fallback: synchronous staging, barrier-per-iteration.
    auto stage = [&](int kt, int s) {
        const __bf16* ga = gA0 + (size_t)kt * 32;
        const __bf16* gb = gB0 + (size_t)kt * 32;
        bf16x8 a0 = ((const bf16x8*)ga)[0];
        bf16x8 a1 = ((const bf16x8*)ga)[1];
        bf16x8 a2 = ((const bf16x8*)ga)[2];
        bf16x8 a3 = ((const bf16x8*)ga)[3];
        bf16x8 b0 = ((const bf16x8*)gb)[0];
        bf16x8 b1 = ((const bf16x8*)gb)[1];
        __bf16* da = &lA[s][tid * 32];
        __bf16* db = &lB[s][bRow * 32 + bSeg];
        ((bf16x8*)da)[0] = a0; ((bf16x8*)da)[1] = a1;
        ((bf16x8*)da)[2] = a2; ((bf16x8*)da)[3] = a3;
        ((bf16x8*)db)[0] = b0; ((bf16x8*)db)[1] = b1;
    };
    stage(0, 0);
    __syncthreads();
    for (int kt = 0; kt < KT; ++kt) {
        int cur = kt % STAGES;
        if (kt + 1 < KT) stage(kt + 1, (kt + 1) % STAGES);
        v16bf afrag[4], bfrag[4];
#pragma unroll
        for (int i = 0; i < 4; ++i)
            afrag[i] = load_fragA(&lA[cur][(waveM + i * 16) * 32], 32, l16, half);
#pragma unroll
        for (int j = 0; j < 4; ++j)
            bfrag[j] = load_fragB(&lB[cur][(waveN + j * 16) * 32], 32, l16, half);
#pragma unroll
        for (int i = 0; i < 4; ++i)
#pragma unroll
            for (int j = 0; j < 4; ++j)
                acc[i][j] = wmma_bf16(afrag[i], bfrag[j], acc[i][j]);
        __syncthreads();
    }
#endif

    // ---- fused LoRA expand: acc += temp[m][0:64] x bbT[id][n][0:64]
    int aid = lora_id[M0 / SQ];
    const __bf16* tA = temp + M0 * RANK;
    const __bf16* tB = bbT + ((size_t)aid * OUTQ + N0) * RANK;
#pragma unroll
    for (int rk = 0; rk < 2; ++rk) {
        v16bf afrag[4], bfrag[4];
#pragma unroll
        for (int i = 0; i < 4; ++i)
            afrag[i] = load_fragA(tA + (size_t)(waveM + i * 16) * RANK + rk * 32,
                                  RANK, l16, half);
#pragma unroll
        for (int j = 0; j < 4; ++j)
            bfrag[j] = load_fragB(tB + (size_t)(waveN + j * 16) * RANK + rk * 32,
                                  RANK, l16, half);
#pragma unroll
        for (int i = 0; i < 4; ++i)
#pragma unroll
            for (int j = 0; j < 4; ++j)
                acc[i][j] = wmma_bf16(afrag[i], bfrag[j], acc[i][j]);
    }

    // ---- epilogue: + bias, store f32
#pragma unroll
    for (int i = 0; i < 4; ++i) {
        size_t gm = M0 + waveM + i * 16 + half * 8;
#pragma unroll
        for (int j = 0; j < 4; ++j) {
            size_t gn = N0 + waveN + j * 16 + l16;
            float bs = bias[gn];
            float* op = out + gm * OUTQ + gn;
#pragma unroll
            for (int v = 0; v < 8; ++v)
                op[(size_t)v * OUTQ] = acc[i][j][v] + bs;
        }
    }
}

// ---------------------------------------------------------------- launch
extern "C" void kernel_launch(void* const* d_in, const int* in_sizes, int n_in,
                              void* d_out, int out_size, void* d_ws, size_t ws_size,
                              hipStream_t stream) {
    const float* x       = (const float*)d_in[0];
    const int*   lora_id = (const int*)d_in[1];
    const float* w       = (const float*)d_in[2];
    const float* bias    = (const float*)d_in[3];
    const float* la      = (const float*)d_in[4];
    const float* lb      = (const float*)d_in[5];
    float* out = (float*)d_out;

    char* ws = (char*)d_ws;
    __bf16* xb   = (__bf16*)ws;                                    // 128 MB
    __bf16* wb   = (__bf16*)(ws + (size_t)134217728);              // 32 MB
    __bf16* temp = (__bf16*)(ws + (size_t)134217728 + 33554432);   // 2 MB
    __bf16* abT  = (__bf16*)(ws + (size_t)134217728 + 33554432 + 2097152);
    __bf16* bbT  = (__bf16*)(ws + (size_t)134217728 + 33554432 + 2097152 + 4194304);

    convert_f32_bf16<<<65536, 256, 0, stream>>>(x, xb);   // 64M elems
    convert_f32_bf16<<<16384, 256, 0, stream>>>(w, wb);   // 16M elems
    transpose_a<<<8192, 256, 0, stream>>>(la, abT);       // 2M elems
    transpose_b<<<8192, 256, 0, stream>>>(lb, bbT);       // 2M elems

    lora_shrink<<<MTOT / 128, 256, 0, stream>>>(xb, abT, lora_id, temp);

    dim3 grid(MTOT / TM, OUTQ / TN);
    lora_main<<<grid, 256, 0, stream>>>(xb, wb, bias, temp, bbT, lora_id, out);
}
